// Adder2D_39762807226650
// MI455X (gfx1250) — compile-verified, matrix operand order checked by reading, not run on hardware
//
#include <hip/hip_runtime.h>

// MI455X / gfx1250 adder-net conv: packed-f16 |x-w| on the VALU, reduction over
// K done by v_wmma_f32_16x16x32_f16 against an all-ones B matrix (f32 accum,
// matrix pipe co-executes with VALU). Input halo staged to LDS as f16 with
// zero padding and a bank-conflict-free channel stride.

typedef __attribute__((ext_vector_type(16))) _Float16 v16h;
typedef __attribute__((ext_vector_type(8)))  _Float16 v8h;
typedef __attribute__((ext_vector_type(8)))  float    v8f;
typedef __attribute__((ext_vector_type(8)))  short    v8s;

#define C_IN   128
#define F_OUT  128
#define KDIM   1152      // 3*3*128
#define HW     32
#define NIMG   32
#define TROWS  6         // 4 output rows + 1 halo row each side
#define TCOLS  34        // 32 cols + 1 halo col each side
#define TCH    136       // padded channel stride (halves) -> conflict-free DS b128

__device__ __forceinline__ v8h habs8(v8h d) {
    // |x| for packed f16 via sign-bit mask: one v_and_b32 per VGPR (2 halves).
    v8s s = __builtin_bit_cast(v8s, d);
    s = s & (short)0x7FFF;
    return __builtin_bit_cast(v8h, s);
}

// Prep: W_col[k,f] (f32, [K][F]) -> wf16[f][k] (f16) so per-lane weight groups
// are single aligned b128 loads in the main loop.
__global__ __launch_bounds__(256) void wcvt_kernel(const float* __restrict__ kern,
                                                   _Float16* __restrict__ wf) {
    int t = blockIdx.x * 256 + threadIdx.x;
    if (t < KDIM * F_OUT) {
        int f = t / KDIM;
        int k = t - f * KDIM;
        wf[t] = (_Float16)kern[k * F_OUT + f];
    }
}

__global__ __launch_bounds__(256) void adder2d_kernel(
        const float* __restrict__ inp,
        const _Float16* __restrict__ wf,
        const float* __restrict__ bias,
        float* __restrict__ outp) {
    __shared__ _Float16 tile[TROWS * TCOLS * TCH];   // 55,488 bytes

    const int tid = threadIdx.x;
    const int n   = blockIdx.x >> 3;          // image
    const int h0  = (blockIdx.x & 7) * 4;     // first of 4 output rows

    // ---- Stage zero-padded halo tile (rows h0-1..h0+4, cols -1..32) as f16 ----
    for (int idx = tid; idx < TROWS * TCOLS * C_IN; idx += 256) {
        int ry = idx / (TCOLS * C_IN);
        int r  = idx - ry * (TCOLS * C_IN);
        int cx = r / C_IN;
        int c  = r - cx * C_IN;
        int gr = h0 - 1 + ry;
        int gc = cx - 1;
        float v = 0.0f;
        if ((unsigned)gr < (unsigned)HW && (unsigned)gc < (unsigned)HW)
            v = inp[((n * HW + gr) * HW + gc) * C_IN + c];
        tile[(ry * TCOLS + cx) * TCH + c] = (_Float16)v;
    }
    __syncthreads();

    const int wid  = tid >> 5;                // 8 wave32s
    const int lane = tid & 31;
    const int hoff = wid >> 1;                // 0..3 -> output row
    const int h    = h0 + hoff;
    const int w0   = (wid & 1) * 16;          // 16-column M-tile base
    const int mcol = w0 + (lane & 15);        // A-matrix row M = lane&15
    const int hh   = lane >> 4;               // lane-half selects K subgroup

    v16h bones;
#pragma unroll
    for (int i = 0; i < 16; ++i) bones[i] = (_Float16)1.0f;

    for (int ft = 0; ft < F_OUT / 4; ++ft) {  // 4 filters per pass
        const int fb = ft * 4;
        v8f acc[4] = {{}, {}, {}, {}};

        for (int kc = 0; kc < KDIM / 32; ++kc) {   // 36 chunks of K=32
            const int kbase = kc * 32;
            const int dy  = kbase / 384;           // wave-uniform -> SALU
            const int rem = kbase - dy * 384;
            const int dx  = rem >> 7;
            const int cc  = rem & 127;             // 0,32,64,96

            // A-layout: elems 0..7 = K hh*8+0..7, elems 8..15 = K 16+hh*8+0..7
            const _Float16* xp =
                &tile[((hoff + dy) * TCOLS + (mcol + dx)) * TCH + cc + hh * 8];
            const v8h xA = *(const v8h*)xp;            // ds_load_b128
            const v8h xB = *(const v8h*)(xp + 16);     // ds_load_b128

            const _Float16* wp = wf + fb * KDIM + kbase + hh * 8;
#pragma unroll
            for (int ff = 0; ff < 4; ++ff) {
                const _Float16* wpf = wp + ff * KDIM;
                v8h dA = habs8(xA - *(const v8h*)wpf);         // v_pk_add(neg)+v_and
                v8h dB = habs8(xB - *(const v8h*)(wpf + 16));
                v16h A = __builtin_shufflevector(dA, dB,
                          0,1,2,3,4,5,6,7,8,9,10,11,12,13,14,15);
                acc[ff] = __builtin_amdgcn_wmma_f32_16x16x32_f16(
                    false, A, false, bones, (short)0, acc[ff], false, false);
            }
        }

        // C layout: VGPR r holds M=r (lanes 0-15) / M=r+8 (lanes 16-31); every
        // N column equals the row sum (ones-B), so lane (r) and lane (16+r)
        // each export one row.
        const float4 bv = *(const float4*)(bias + fb);
#pragma unroll
        for (int r = 0; r < 8; ++r) {
            if ((lane & 15) == r) {
                const int col = w0 + r + 8 * hh;
                float4 o;
                o.x = bv.x - acc[0][r];
                o.y = bv.y - acc[1][r];
                o.z = bv.z - acc[2][r];
                o.w = bv.w - acc[3][r];
                *(float4*)(outp + (((n * HW + h) * HW + col) * F_OUT + fb)) = o;
            }
        }
    }
}

extern "C" void kernel_launch(void* const* d_in, const int* in_sizes, int n_in,
                              void* d_out, int out_size, void* d_ws, size_t ws_size,
                              hipStream_t stream) {
    (void)in_sizes; (void)n_in; (void)out_size; (void)ws_size;
    const float* inp  = (const float*)d_in[0];   // [32,32,32,128] f32
    const float* kern = (const float*)d_in[1];   // [3,3,128,128]  f32
    const float* bias = (const float*)d_in[2];   // [128]          f32
    float* outp = (float*)d_out;                 // [32,32,32,128] f32
    _Float16* wf = (_Float16*)d_ws;              // 1152*128 f16 = 288 KB scratch

    wcvt_kernel<<<dim3((KDIM * F_OUT + 255) / 256), dim3(256), 0, stream>>>(kern, wf);
    adder2d_kernel<<<dim3(NIMG * (HW / 4)), dim3(256), 0, stream>>>(inp, wf, bias, outp);
}
